// kernel_generated_0_62904091017429
// MI455X (gfx1250) — compile-verified, matrix operand order checked by reading, not run on hardware
//
#include <hip/hip_runtime.h>

// ---- CDNA5 WMMA types -------------------------------------------------------
typedef __attribute__((ext_vector_type(16))) __bf16 v16bf;
typedef __attribute__((ext_vector_type(8)))  __bf16 v8bf;
typedef __attribute__((ext_vector_type(8)))  float  v8f;

// Problem constants
#define N_BATCH 128
#define C_IN    128
#define HGT     56
#define WID     56
#define O_CH    32
#define KTAP    3
#define KDIM    (KTAP * C_IN)      // 384

// LDS layouts (bank-conflict-free strides: 68 resp. 196 dwords == 4 mod 64)
#define LDSX_STRIDE 136            // ushorts per row of x-tile  (>=128, 16B aligned)
#define LDSX_ROWS   66             // t = 0..65 (covers w<=63 + tap<=2)
#define LDSW_STRIDE 392            // ushorts per row of weights (>=384, 16B aligned)

static __device__ __forceinline__ unsigned short f32_to_bf16(float f) {
    unsigned int u = __builtin_bit_cast(unsigned int, f);
    // round-to-nearest-even truncation to bf16
    u += 0x7FFFu + ((u >> 16) & 1u);
    return (unsigned short)(u >> 16);
}

static __device__ __forceinline__ v16bf load_frag(const unsigned short* p) {
    // two contiguous 16B LDS reads -> ds_load_b128 each
    v8bf lo = *reinterpret_cast<const v8bf*>(p);
    v8bf hi = *reinterpret_cast<const v8bf*>(p + 16);
    return __builtin_shufflevector(lo, hi, 0, 1, 2, 3, 4, 5, 6, 7,
                                           8, 9, 10, 11, 12, 13, 14, 15);
}

__global__ __launch_bounds__(256)
void shiftconv_wmma_bf16(const float* __restrict__ x,
                         const float* __restrict__ wgt,
                         float* __restrict__ out) {
    __shared__ alignas(16) unsigned short lds_x[LDSX_ROWS * LDSX_STRIDE];
    __shared__ alignas(16) unsigned short lds_w[O_CH * LDSW_STRIDE];

    const int tid = threadIdx.x;
    const int n   = blockIdx.x / HGT;    // batch index
    const int h   = blockIdx.x % HGT;    // input H row

    // ---- zero the pad rows of lds_x: row 0 (t'=-1) and rows 57..65 ---------
    for (int idx = tid; idx < 10 * C_IN; idx += 256) {
        int r   = idx >> 7;                    // 0..9
        int c   = idx & 127;
        int row = (r == 0) ? 0 : (56 + r);     // 0, 57..65
        lds_x[row * LDSX_STRIDE + c] = 0;
    }

    // ---- stage rolled x row (transposed, bf16): lds_x[t'+1][i] = xr[i][t'] --
    // xr[t'] = x[(t'-1) mod 56]  =>  source w lands at row ((w+1)%56)+1
    const float* xrow = x + (size_t)n * C_IN * HGT * WID + (size_t)h * WID;
    for (int idx = tid; idx < C_IN * WID; idx += 256) {
        int i = idx / WID;
        int w = idx % WID;
        float v = xrow[(size_t)i * (HGT * WID) + w];
        int t = w + 1; if (t >= WID) t -= WID;
        lds_x[(t + 1) * LDSX_STRIDE + i] = f32_to_bf16(v);
    }

    // ---- stage weights (bf16), K ordering kk = k*128 + i (matches w flat) --
    for (int idx = tid; idx < O_CH * KDIM; idx += 256) {
        int j = idx / KDIM;
        int r = idx % KDIM;                    // r = k*128 + i
        lds_w[j * LDSW_STRIDE + r] = f32_to_bf16(wgt[idx]);
    }

    __syncthreads();

    // ---- per-wave tile: 2 M-tiles x 4 N-tiles --------------------------------
    const int lane  = tid & 31;
    const int wv    = tid >> 5;                // 0..7
    const int m0    = (wv & 1) * 16;           // output-channel tile base
    const int w0    = (wv >> 1) * 16;          // W tile base (0,16,32,48)
    const int l16   = lane & 15;
    const int lhalf = lane >> 4;               // 0 or 1

    // A fragment base: row j = m0 + l16; per step s: K = s*32 + lhalf*8 (+16)
    const unsigned short* pa_base = &lds_w[(m0 + l16) * LDSW_STRIDE + lhalf * 8];
    const int wcol = w0 + l16;                 // B column (output w)

    v8f acc = {};
#pragma unroll
    for (int s = 0; s < KDIM / 32; ++s) {      // 12 K-steps
        v16bf a = load_frag(pa_base + s * 32);
        // kk = s*32 + lhalf*8 + e  ->  tap = s/4, i = (s&3)*32 + lhalf*8 + e
        const int tap   = s >> 2;
        const int ibase = (s & 3) * 32 + lhalf * 8;
        v16bf b = load_frag(&lds_x[(wcol + tap) * LDSX_STRIDE + ibase]);
        acc = __builtin_amdgcn_wmma_f32_16x16x32_bf16(
            /*neg_a=*/false, a, /*neg_b=*/false, b,
            /*c_mod=*/(short)0, acc, /*reuse_a=*/false, /*reuse_b=*/false);
    }

    // ---- epilogue: apply output H-roll, store valid columns -----------------
    // C layout: lanes 0-15 -> N=lane, M=r ; lanes 16-31 -> N=lane-16, M=r+8
    int h_out = h + 1; if (h_out >= HGT) h_out = 0;
    if (wcol < WID) {
#pragma unroll
        for (int r = 0; r < 8; ++r) {
            int j = m0 + r + lhalf * 8;
            out[(((size_t)n * O_CH + j) * HGT + h_out) * WID + wcol] = acc[r];
        }
    }
}

extern "C" void kernel_launch(void* const* d_in, const int* in_sizes, int n_in,
                              void* d_out, int out_size, void* d_ws, size_t ws_size,
                              hipStream_t stream) {
    (void)in_sizes; (void)n_in; (void)out_size; (void)d_ws; (void)ws_size;
    const float* x   = (const float*)d_in[0];   // (128,128,56,56) f32
    const float* w   = (const float*)d_in[1];   // (32,3,128) f32
    float*       out = (float*)d_out;           // (128,32,56,56) f32

    dim3 grid(N_BATCH * HGT);                   // one block per (n, h) row
    dim3 block(256);                            // 8 wave32 = 2 M-tiles x 4 N-tiles
    shiftconv_wmma_bf16<<<grid, block, 0, stream>>>(x, w, out);
}
